// GNN_node_Virtualnode_73710228734482
// MI455X (gfx1250) — matrix operand order repaired and checked by reading, not compiled, not run on hardware
//
#include <hip/hip_runtime.h>
#include <hip/hip_bf16.h>

// Problem constants (match reference)
#define NN 50000
#define NE 800000
#define DDIM 128
#define HDIM 256
#define NG 512
#define NL 3
#define BN_EPS 1e-5f

typedef __attribute__((ext_vector_type(16))) __bf16 v16bf;
typedef __attribute__((ext_vector_type(8)))  __bf16 v8bf;
typedef __attribute__((ext_vector_type(8)))  float  v8f;

// ---------------------------------------------------------------------------
// h_in = h + vn[batch]
// ---------------------------------------------------------------------------
__global__ __launch_bounds__(256)
void hin_kernel(const float* __restrict__ h, const float* __restrict__ vn,
                const int* __restrict__ batch, float* __restrict__ h_in, int n)
{
    int i = blockIdx.x * 256 + threadIdx.x;
    if (i >= n) return;
    int node = i >> 7;          // D = 128
    int c    = i & 127;
    h_in[i] = h[i] + vn[(size_t)batch[node] * DDIM + c];
}

// ---------------------------------------------------------------------------
// vn init: broadcast vn_emb[0,:] to all G graphs
// ---------------------------------------------------------------------------
__global__ __launch_bounds__(256)
void vninit_kernel(const float* __restrict__ vn_emb, float* __restrict__ vn, int n)
{
    int i = blockIdx.x * 256 + threadIdx.x;
    if (i >= n) return;
    vn[i] = vn_emb[i & 127];
}

// ---------------------------------------------------------------------------
// Fused edge encoder + GIN message + scatter-add:
//   e   = edge_attr @ We + be        (8x128 weights held in LDS)
//   msg = relu(h_in[src] + e)
//   aggr[dst] += msg                 (native f32 global atomics)
// One wave per edge; each lane owns 4 consecutive channels.
// ---------------------------------------------------------------------------
__global__ __launch_bounds__(256)
void edge_kernel(const float* __restrict__ edge_attr,
                 const int*   __restrict__ edge_index,
                 const float* __restrict__ We,   // [8,128] layer slice
                 const float* __restrict__ be,   // [128]
                 const float* __restrict__ h_in,
                 float*       __restrict__ aggr,
                 int nE)
{
    __shared__ float sW[8 * DDIM];
    __shared__ float sb[DDIM];
    int tid = threadIdx.x;
    for (int i = tid; i < 8 * DDIM; i += 256) sW[i] = We[i];
    if (tid < DDIM) sb[tid] = be[tid];
    __syncthreads();

    int e = blockIdx.x * 8 + (tid >> 5);
    if (e >= nE) return;
    int lane = tid & 31;
    int c0   = lane * 4;

    float ea[8];
#pragma unroll
    for (int k = 0; k < 8; ++k) ea[k] = edge_attr[(size_t)e * 8 + k];

    float a0 = sb[c0 + 0], a1 = sb[c0 + 1], a2 = sb[c0 + 2], a3 = sb[c0 + 3];
#pragma unroll
    for (int k = 0; k < 8; ++k) {
        const float* wr = &sW[k * DDIM + c0];
        a0 = fmaf(ea[k], wr[0], a0);
        a1 = fmaf(ea[k], wr[1], a1);
        a2 = fmaf(ea[k], wr[2], a2);
        a3 = fmaf(ea[k], wr[3], a3);
    }

    int s = edge_index[e];
    int d = edge_index[nE + e];
    const float4 hv = *(const float4*)&h_in[(size_t)s * DDIM + c0];
    float m0 = fmaxf(hv.x + a0, 0.0f);
    float m1 = fmaxf(hv.y + a1, 0.0f);
    float m2 = fmaxf(hv.z + a2, 0.0f);
    float m3 = fmaxf(hv.w + a3, 0.0f);
    float* ap = &aggr[(size_t)d * DDIM + c0];
    unsafeAtomicAdd(ap + 0, m0);
    unsafeAtomicAdd(ap + 1, m1);
    unsafeAtomicAdd(ap + 2, m2);
    unsafeAtomicAdd(ap + 3, m3);
}

// ---------------------------------------------------------------------------
// pooled += h_in per graph (batch is sorted; values mostly hit L2)
// ---------------------------------------------------------------------------
__global__ __launch_bounds__(256)
void pool_kernel(const float* __restrict__ h_in, const int* __restrict__ batch,
                 float* __restrict__ pooled, int nNodes)
{
    int i = blockIdx.x * 256 + threadIdx.x;
    int node = i >> 5;
    if (node >= nNodes) return;
    int c0 = (i & 31) * 4;
    int g  = batch[node];
    const float4 v = *(const float4*)&h_in[(size_t)node * DDIM + c0];
    float* p = &pooled[(size_t)g * DDIM + c0];
    unsafeAtomicAdd(p + 0, v.x);
    unsafeAtomicAdd(p + 1, v.y);
    unsafeAtomicAdd(p + 2, v.z);
    unsafeAtomicAdd(p + 3, v.w);
}

// ---------------------------------------------------------------------------
// Convert fp32 weights [K, Ncol] into WMMA-fragment-major bf16:
// frag f = kt*(Ncol/16)+nt; per lane L (0..31), 16 contiguous bf16:
//   col = nt*16 + (L&15),  k = kt*32 + 16*(L>=16) + j        (contiguous k order, same as A side)
// ---------------------------------------------------------------------------
__global__ __launch_bounds__(256)
void wfrag_kernel(const float* __restrict__ W, __bf16* __restrict__ Wfrag,
                  int K, int Ncol)
{
    int i = blockIdx.x * 256 + threadIdx.x;
    int total = K * Ncol;
    if (i >= total) return;
    int j  = i & 15;
    int L  = (i >> 4) & 31;
    int f  = i >> 9;
    int nfrags = Ncol >> 4;
    int kt = f / nfrags;
    int nt = f - kt * nfrags;
    int col = nt * 16 + (L & 15);
    int k   = kt * 32 + ((L & 16) ? 16 : 0) + j;
    Wfrag[i] = (__bf16)W[(size_t)k * Ncol + col];
}

// ---------------------------------------------------------------------------
// WMMA GEMM with fused A-side transform + fused BN-stat accumulation:
//   A = A0 (+ A1)                      (z = h_in + aggr)
//   A = relu(affA*A + affC)            (fused BN+relu of previous stage)
//   P = A @ W + bias ;  stats[col] += sum, stats[Ncol+col] += sum of squares
// Block: 256 threads = 8 waves; block tile 128 rows x 128 cols;
// wave w owns rows [w*16, w*16+16) x 128 cols = 8 v8f accumulators.
// ---------------------------------------------------------------------------
__global__ __launch_bounds__(256)
void gemm_bn_kernel(const float* __restrict__ A0,
                    const float* __restrict__ A1,
                    const float* __restrict__ affA,
                    const float* __restrict__ affC,
                    const __bf16* __restrict__ Wfrag,
                    const float* __restrict__ bias,
                    float* __restrict__ P,
                    float* __restrict__ stats,
                    int M, int K, int Ncol)
{
    __shared__ __bf16 Atile[128][32];   // 8 KB
    __shared__ float  s_sum[128];
    __shared__ float  s_ssq[128];

    const int tid  = threadIdx.x;
    const int lane = tid & 31;
    const int wave = tid >> 5;
    const int row0 = blockIdx.x * 128;
    const int col0 = blockIdx.y * 128;

    if (tid < 128) { s_sum[tid] = 0.0f; s_ssq[tid] = 0.0f; }

    v8f acc[8];
#pragma unroll
    for (int i = 0; i < 8; ++i)
#pragma unroll
        for (int r = 0; r < 8; ++r) acc[i][r] = 0.0f;

    const int lr   = tid >> 1;          // A-tile row this thread loads (0..127)
    const int cseg = (tid & 1) * 16;    // 16-column segment
    const int grow = row0 + lr;

    // A fragment addressing (contiguous k order, matches Wfrag): lane half selects K base
    const int lrw = wave * 16 + (lane & 15);
    const int kb  = (lane & 16) ? 16 : 0;

    const int ktiles = K >> 5;
    const int ntile0 = col0 >> 4;
    const int nfrags = Ncol >> 4;

    for (int kt = 0; kt < ktiles; ++kt) {
        float vals[16];
        const int kcol = kt * 32 + cseg;
        if (grow < M) {
            const float* s0 = A0 + (size_t)grow * K + kcol;
#pragma unroll
            for (int j = 0; j < 16; ++j) vals[j] = s0[j];
            if (A1 != nullptr) {
                const float* s1 = A1 + (size_t)grow * K + kcol;
#pragma unroll
                for (int j = 0; j < 16; ++j) vals[j] += s1[j];
            }
            if (affA != nullptr) {
#pragma unroll
                for (int j = 0; j < 16; ++j) {
                    int c = kcol + j;
                    vals[j] = fmaxf(fmaf(affA[c], vals[j], affC[c]), 0.0f);
                }
            }
        } else {
#pragma unroll
            for (int j = 0; j < 16; ++j) vals[j] = 0.0f;
        }
        __syncthreads();   // previous iteration's fragment reads done
#pragma unroll
        for (int j = 0; j < 16; ++j) Atile[lr][cseg + j] = (__bf16)vals[j];
        __syncthreads();

        const v8bf alo = *(const v8bf*)&Atile[lrw][kb];        // K = kb..kb+7
        const v8bf ahi = *(const v8bf*)&Atile[lrw][kb + 8];    // K = kb+8..kb+15
        const v16bf a = __builtin_shufflevector(
            alo, ahi, 0, 1, 2, 3, 4, 5, 6, 7, 8, 9, 10, 11, 12, 13, 14, 15);

#pragma unroll
        for (int nt = 0; nt < 8; ++nt) {
            const size_t f  = (size_t)kt * nfrags + (ntile0 + nt);
            const v8bf* bp  = (const v8bf*)(Wfrag + (f * 32 + lane) * 16);
            const v16bf b = __builtin_shufflevector(
                bp[0], bp[1], 0, 1, 2, 3, 4, 5, 6, 7, 8, 9, 10, 11, 12, 13, 14, 15);
            acc[nt] = __builtin_amdgcn_wmma_f32_16x16x32_bf16(
                false, a, false, b, (short)0, acc[nt], false, false);
        }
    }

    // Epilogue: bias, store, BN stats (D layout: row = r + 8*(lane>=16), col = lane%16)
    const int rbase = row0 + wave * 16 + ((lane & 16) ? 8 : 0);
#pragma unroll
    for (int nt = 0; nt < 8; ++nt) {
        const int lcol = nt * 16 + (lane & 15);
        const int col  = col0 + lcol;
        const float bv = (bias != nullptr) ? bias[col] : 0.0f;
        float lsum = 0.0f, lssq = 0.0f;
#pragma unroll
        for (int r = 0; r < 8; ++r) {
            const int row = rbase + r;
            if (row < M) {
                const float v = acc[nt][r] + bv;
                P[(size_t)row * Ncol + col] = v;
                lsum += v;
                lssq += v * v;
            }
        }
        atomicAdd(&s_sum[lcol], lsum);
        atomicAdd(&s_ssq[lcol], lssq);
    }
    __syncthreads();
    if (tid < 128) {
        unsafeAtomicAdd(&stats[col0 + tid], s_sum[tid]);
        unsafeAtomicAdd(&stats[Ncol + col0 + tid], s_ssq[tid]);
    }
}

// ---------------------------------------------------------------------------
// BN finalize: stats -> per-column affine  a = g*rsqrt(var+eps), c = beta - mean*a
// ---------------------------------------------------------------------------
__global__ __launch_bounds__(256)
void bnfin_kernel(const float* __restrict__ stats, const float* __restrict__ g,
                  const float* __restrict__ beta, float* __restrict__ a,
                  float* __restrict__ c, int Ncol, float invM)
{
    int i = blockIdx.x * 256 + threadIdx.x;
    if (i >= Ncol) return;
    float mean = stats[i] * invM;
    float var  = stats[Ncol + i] * invM - mean * mean;
    float s    = g[i] * rsqrtf(var + BN_EPS);
    a[i] = s;
    c[i] = beta[i] - mean * s;
}

// ---------------------------------------------------------------------------
// y = (relu?) (a[col]*x + c[col])    (cols is a power of two: 128)
// ---------------------------------------------------------------------------
__global__ __launch_bounds__(256)
void affine_kernel(const float* __restrict__ x, const float* __restrict__ a,
                   const float* __restrict__ c, float* __restrict__ y,
                   int n, int colmask, int relu)
{
    int i = blockIdx.x * 256 + threadIdx.x;
    if (i >= n) return;
    int col = i & colmask;
    float v = fmaf(a[col], x[i], c[col]);
    if (relu) v = fmaxf(v, 0.0f);
    y[i] = v;
}

// ===========================================================================
extern "C" void kernel_launch(void* const* d_in, const int* in_sizes, int n_in,
                              void* d_out, int out_size, void* d_ws, size_t ws_size,
                              hipStream_t stream)
{
    const float* x        = (const float*)d_in[0];
    const float* edgeattr = (const float*)d_in[1];
    const float* vn_emb   = (const float*)d_in[2];
    const float* We       = (const float*)d_in[3];
    const float* be       = (const float*)d_in[4];
    const float* W1       = (const float*)d_in[5];
    const float* b1       = (const float*)d_in[6];
    const float* g1       = (const float*)d_in[7];
    const float* bt1      = (const float*)d_in[8];
    const float* W2       = (const float*)d_in[9];
    const float* b2       = (const float*)d_in[10];
    const float* gb       = (const float*)d_in[11];
    const float* bb       = (const float*)d_in[12];
    const float* Wv1      = (const float*)d_in[13];
    const float* bv1      = (const float*)d_in[14];
    const float* gv1      = (const float*)d_in[15];
    const float* btv1     = (const float*)d_in[16];
    const float* Wv2      = (const float*)d_in[17];
    const float* bv2      = (const float*)d_in[18];
    const float* gv2      = (const float*)d_in[19];
    const float* btv2     = (const float*)d_in[20];
    const int*   eidx     = (const int*)d_in[21];
    const int*   batch    = (const int*)d_in[22];
    float*       out      = (float*)d_out;

    // ---- workspace carving (256B aligned) ----
    char* ws = (char*)d_ws;
    size_t off = 0;
    auto take = [&](size_t bytes) -> char* {
        char* p = ws + off;
        off += (bytes + 255) & ~(size_t)255;
        return p;
    };
    float*  h_buf  = (float*)take((size_t)NN * DDIM * 4);
    float*  h_in   = (float*)take((size_t)NN * DDIM * 4);
    float*  aggr   = (float*)take((size_t)NN * DDIM * 4);
    float*  pre1   = (float*)take((size_t)NN * HDIM * 4);
    float*  pre2   = (float*)take((size_t)NN * DDIM * 4);
    float*  vn     = (float*)take((size_t)NG * DDIM * 4);
    float*  pooled = (float*)take((size_t)NG * DDIM * 4);
    float*  pv1    = (float*)take((size_t)NG * HDIM * 4);
    float*  pv2    = (float*)take((size_t)NG * DDIM * 4);
    __bf16* Wb1    = (__bf16*)take((size_t)DDIM * HDIM * 2);
    __bf16* Wb2    = (__bf16*)take((size_t)HDIM * DDIM * 2);
    __bf16* Wbv1   = (__bf16*)take((size_t)DDIM * HDIM * 2);
    __bf16* Wbv2   = (__bf16*)take((size_t)HDIM * DDIM * 2);
    float*  stats  = (float*)take(2 * HDIM * 4);
    float*  a1     = (float*)take(HDIM * 4);
    float*  c1     = (float*)take(HDIM * 4);
    float*  a2     = (float*)take(DDIM * 4);
    float*  c2     = (float*)take(DDIM * 4);
    float*  av1    = (float*)take(HDIM * 4);
    float*  cv1    = (float*)take(HDIM * 4);
    float*  av2    = (float*)take(DDIM * 4);
    float*  cv2    = (float*)take(DDIM * 4);
    if (off > ws_size) return;   // workspace too small; bail deterministically

    const int rowTiles  = (NN + 127) / 128;          // 391
    const int gTiles    = (NG + 127) / 128;          // 4
    const float invN    = 1.0f / (float)NN;
    const float invG    = 1.0f / (float)NG;

    // vn <- broadcast(vn_emb)
    vninit_kernel<<<(NG * DDIM) / 256, 256, 0, stream>>>(vn_emb, vn, NG * DDIM);

    const float* h_cur = x;
    for (int l = 0; l < NL; ++l) {
        // h_in = h + vn[batch]
        hin_kernel<<<(NN * DDIM) / 256, 256, 0, stream>>>(h_cur, vn, batch, h_in, NN * DDIM);

        // edge phase: aggr = segment_sum(relu(h_in[src] + edge_attr@We + be), dst)
        hipMemsetAsync(aggr, 0, (size_t)NN * DDIM * 4, stream);
        edge_kernel<<<NE / 8, 256, 0, stream>>>(
            edgeattr, eidx, We + (size_t)l * 8 * DDIM, be + (size_t)l * DDIM,
            h_in, aggr, NE);

        // GEMM1: pre1 = (h_in + aggr) @ W1[l] + b1[l]   (+ column stats)
        wfrag_kernel<<<(DDIM * HDIM) / 256, 256, 0, stream>>>(
            W1 + (size_t)l * DDIM * HDIM, Wb1, DDIM, HDIM);
        hipMemsetAsync(stats, 0, 2 * HDIM * 4, stream);
        gemm_bn_kernel<<<dim3(rowTiles, HDIM / 128), 256, 0, stream>>>(
            h_in, aggr, nullptr, nullptr, Wb1, b1 + (size_t)l * HDIM,
            pre1, stats, NN, DDIM, HDIM);
        bnfin_kernel<<<1, 256, 0, stream>>>(
            stats, g1 + (size_t)l * HDIM, bt1 + (size_t)l * HDIM, a1, c1, HDIM, invN);

        // GEMM2: pre2 = relu(BN(pre1)) @ W2[l] + b2[l]  (BN+relu fused into A-load)
        wfrag_kernel<<<(HDIM * DDIM) / 256, 256, 0, stream>>>(
            W2 + (size_t)l * HDIM * DDIM, Wb2, HDIM, DDIM);
        hipMemsetAsync(stats, 0, 2 * DDIM * 4, stream);
        gemm_bn_kernel<<<dim3(rowTiles, DDIM / 128), 256, 0, stream>>>(
            pre1, nullptr, a1, c1, Wb2, b2 + (size_t)l * DDIM,
            pre2, stats, NN, HDIM, DDIM);
        bnfin_kernel<<<1, 256, 0, stream>>>(
            stats, gb + (size_t)l * DDIM, bb + (size_t)l * DDIM, a2, c2, DDIM, invN);

        if (l < NL - 1) {
            // h = relu(BN(pre2))
            affine_kernel<<<(NN * DDIM) / 256, 256, 0, stream>>>(
                pre2, a2, c2, h_buf, NN * DDIM, DDIM - 1, 1);

            // pooled = segment_sum(h_in, batch); vn MLP
            hipMemsetAsync(pooled, 0, (size_t)NG * DDIM * 4, stream);
            pool_kernel<<<(NN * 32) / 256, 256, 0, stream>>>(h_in, batch, pooled, NN);

            wfrag_kernel<<<(DDIM * HDIM) / 256, 256, 0, stream>>>(
                Wv1 + (size_t)l * DDIM * HDIM, Wbv1, DDIM, HDIM);
            hipMemsetAsync(stats, 0, 2 * HDIM * 4, stream);
            gemm_bn_kernel<<<dim3(gTiles, HDIM / 128), 256, 0, stream>>>(
                pooled, vn, nullptr, nullptr, Wbv1, bv1 + (size_t)l * HDIM,
                pv1, stats, NG, DDIM, HDIM);
            bnfin_kernel<<<1, 256, 0, stream>>>(
                stats, gv1 + (size_t)l * HDIM, btv1 + (size_t)l * HDIM, av1, cv1, HDIM, invG);

            wfrag_kernel<<<(HDIM * DDIM) / 256, 256, 0, stream>>>(
                Wv2 + (size_t)l * HDIM * DDIM, Wbv2, HDIM, DDIM);
            hipMemsetAsync(stats, 0, 2 * DDIM * 4, stream);
            gemm_bn_kernel<<<dim3(gTiles, DDIM / 128), 256, 0, stream>>>(
                pv1, nullptr, av1, cv1, Wbv2, bv2 + (size_t)l * DDIM,
                pv2, stats, NG, HDIM, DDIM);
            bnfin_kernel<<<1, 256, 0, stream>>>(
                stats, gv2 + (size_t)l * DDIM, btv2 + (size_t)l * DDIM, av2, cv2, DDIM, invG);

            // vn = relu(BN(pv2))
            affine_kernel<<<(NG * DDIM) / 256, 256, 0, stream>>>(
                pv2, av2, cv2, vn, NG * DDIM, DDIM - 1, 1);

            h_cur = h_buf;
        } else {
            // final output: h = BN(pre2), no relu
            affine_kernel<<<(NN * DDIM) / 256, 256, 0, stream>>>(
                pre2, a2, c2, out, NN * DDIM, DDIM - 1, 0);
        }
    }
}
